// SpatialNCA_27238682591241
// MI455X (gfx1250) — compile-verified, matrix operand order checked by reading, not run on hardware
//
#include <hip/hip_runtime.h>

#define NN 50000
#define DD 64
#define EE 800000

typedef __attribute__((ext_vector_type(16))) _Float16 v16h;
typedef __attribute__((ext_vector_type(8)))  _Float16 v8h;
typedef __attribute__((ext_vector_type(2)))  _Float16 v2h;
typedef __attribute__((ext_vector_type(8)))  float    v8f;

__device__ __forceinline__ v8f wmma16x16x32(v16h a, v16h b, v8f c) {
  // D = A(16x32,f16) * B(32x16,f16) + C(16x16,f32)
  return __builtin_amdgcn_wmma_f32_16x16x32_f16(false, a, false, b, (short)0, c,
                                                false, false);
}

// v_cvt_pk_rtz_f16_f32 wrapper (builtin returns __fp16 vector; bit-cast it).
__device__ __forceinline__ v2h pk2(float x, float y) {
  return __builtin_bit_cast(v2h, __builtin_amdgcn_cvt_pkrtz(x, y));
}

// K index inside a 32-wide chunk for f16 A/B fragments (ISA 7.12.2):
// lanes 0-15 hold K in {0..7, 16..23}, lanes 16-31 hold K in {8..15, 24..31}.
__device__ __forceinline__ int kmap(int e, int laneHi) {
  return (e & 7) + ((e >> 3) << 4) + (laneHi << 3);
}

// Weight fragment (A operand = W^T tile) from fragment-order packed LDS:
// packed[(chunk*32+lane)*16 + e] == W[kt*32+kmap(e,laneHi)][ft*16 + (lane&15)]
// One contiguous 32B read per lane -> 2x ds_load_b128.
__device__ __forceinline__ v16h load_fragW(const _Float16* packed, int chunk,
                                           int lane) {
  return *(const v16h*)&packed[(chunk * 32 + lane) * 16];
}

// Swizzle a row-major [K x 64] f32 weight matrix into fragment order (f16).
// kchunks = K/32, chunk = kt*4 + ft.
__device__ __forceinline__ void pack_weight(const float* __restrict__ W,
                                            _Float16* __restrict__ packed,
                                            int kchunks, int tid,
                                            int nthreads) {
  const int total = kchunks * 4 * 32 * 16;
  for (int i = tid; i < total; i += nthreads) {
    const int e = i & 15;
    const int lane = (i >> 4) & 31;
    const int chunk = i >> 9;
    const int kt = chunk >> 2, ft = chunk & 3;
    const int laneHi = lane >> 4, col = lane & 15;
    packed[i] = (_Float16)W[(kt * 32 + kmap(e, laneHi)) * DD + ft * 16 + col];
  }
}

// Build a B fragment (one 32-K chunk) from four float4s via packed cvt.
__device__ __forceinline__ v16h pack16(const float4 a, const float4 b,
                                       const float4 c, const float4 d) {
  const v2h p0 = pk2(a.x, a.y);
  const v2h p1 = pk2(a.z, a.w);
  const v2h p2 = pk2(b.x, b.y);
  const v2h p3 = pk2(b.z, b.w);
  const v2h p4 = pk2(c.x, c.y);
  const v2h p5 = pk2(c.z, c.w);
  const v2h p6 = pk2(d.x, d.y);
  const v2h p7 = pk2(d.z, d.w);
  v16h r;
  r[0] = p0[0]; r[1] = p0[1]; r[2] = p1[0]; r[3] = p1[1];
  r[4] = p2[0]; r[5] = p2[1]; r[6] = p3[0]; r[7] = p3[1];
  r[8] = p4[0]; r[9] = p4[1]; r[10] = p5[0]; r[11] = p5[1];
  r[12] = p6[0]; r[13] = p6[1]; r[14] = p7[0]; r[15] = p7[1];
  return r;
}

// Build next-GEMM B fragment chunk kt from D tiles 2kt (t0) and 2kt+1 (t1):
// element e<8 -> t0[e], e>=8 -> t1[e-8]  (layout identity, in-register).
__device__ __forceinline__ v16h pack16_acc(const v8f t0, const v8f t1) {
  const v2h p0 = pk2(t0[0], t0[1]);
  const v2h p1 = pk2(t0[2], t0[3]);
  const v2h p2 = pk2(t0[4], t0[5]);
  const v2h p3 = pk2(t0[6], t0[7]);
  const v2h p4 = pk2(t1[0], t1[1]);
  const v2h p5 = pk2(t1[2], t1[3]);
  const v2h p6 = pk2(t1[4], t1[5]);
  const v2h p7 = pk2(t1[6], t1[7]);
  v16h r;
  r[0] = p0[0]; r[1] = p0[1]; r[2] = p1[0]; r[3] = p1[1];
  r[4] = p2[0]; r[5] = p2[1]; r[6] = p3[0]; r[7] = p3[1];
  r[8] = p4[0]; r[9] = p4[1]; r[10] = p5[0]; r[11] = p5[1];
  r[12] = p6[0]; r[13] = p6[1]; r[14] = p7[0]; r[15] = p7[1];
  return r;
}

__device__ __forceinline__ float silu_f(float x) {
  return x * __builtin_amdgcn_rcpf(1.0f + __expf(-x));
}

// ---------------------------------------------------------------------------
// prep: h_eff = h + h_init ; zero agg / pos_acc / deg
// ---------------------------------------------------------------------------
__global__ void egnn_prep_kernel(const float* __restrict__ h,
                                 const float* __restrict__ h_init,
                                 float* __restrict__ h_eff,
                                 float* __restrict__ agg,
                                 float* __restrict__ pos_acc,
                                 float* __restrict__ deg) {
  const int i = blockIdx.x * 256 + threadIdx.x;
  if (i < NN * DD) { h_eff[i] = h[i] + h_init[i]; agg[i] = 0.0f; }
  if (i < NN * 2) pos_acc[i] = 0.0f;
  if (i < NN) deg[i] = 0.0f;
}

// ---------------------------------------------------------------------------
// edge kernel (fully transposed chain, lane = edge, no activation staging)
// ---------------------------------------------------------------------------
__global__ __launch_bounds__(256) void egnn_edge_kernel(
    const float* __restrict__ h_eff, const float* __restrict__ pos,
    const int* __restrict__ ei,
    const float* __restrict__ W_msg1, const float* __restrict__ b_msg1,
    const float* __restrict__ W_msg2, const float* __restrict__ b_msg2,
    const float* __restrict__ W_pos1, const float* __restrict__ b_pos1,
    const float* __restrict__ W_pos2, const float* __restrict__ b_pos2,
    float* __restrict__ agg, float* __restrict__ pos_acc,
    float* __restrict__ deg) {
  __shared__ __align__(32) _Float16 sW1p[16 * 32 * 16];  // W_msg1 rows 0..127
  __shared__ __align__(32) _Float16 sW2p[8 * 32 * 16];   // W_msg2
  __shared__ __align__(32) _Float16 sWp1p[8 * 32 * 16];  // W_pos1
  __shared__ __align__(16) float sW1d2[DD];              // W_msg1 row 128 (d2)
  __shared__ __align__(16) float sb1[DD], sb2[DD], sbp1[DD], sWp2[DD];
  __shared__ float sbp2;

  const int tid = threadIdx.x;
  pack_weight(W_msg1, sW1p, 4, tid, 256);
  pack_weight(W_msg2, sW2p, 2, tid, 256);
  pack_weight(W_pos1, sWp1p, 2, tid, 256);
  if (tid < DD) {
    sW1d2[tid] = W_msg1[128 * DD + tid];
    sb1[tid]  = b_msg1[tid];
    sb2[tid]  = b_msg2[tid];
    sbp1[tid] = b_pos1[tid];
    sWp2[tid] = W_pos2[tid];
  }
  if (tid == 0) sbp2 = b_pos2[0];
  __syncthreads();

  const int wave = tid >> 5, lane = tid & 31;
  const int laneHi = lane >> 4, lr = lane & 15;
  const int edge = blockIdx.x * 128 + wave * 16 + lr;
  const int s = ei[edge];
  const int d = ei[EE + edge];
  const float* __restrict__ hd = &h_eff[(size_t)d * DD];
  const float* __restrict__ hs = &h_eff[(size_t)s * DD];
  const float r0 = pos[d * 2 + 0] - pos[s * 2 + 0];
  const float r1 = pos[d * 2 + 1] - pos[s * 2 + 1];
  const float d2v = r0 * r0 + r1 * r1;

  const v8f z = {};

  // GEMM1: D1 = W1^T @ [h[dst]|h[src]]^T  (lane = edge column)
  v8f acc[4] = {z, z, z, z};
#pragma unroll
  for (int kt = 0; kt < 4; ++kt) {
    const float4* rp = (const float4*)((kt < 2) ? hd : hs);
    const int i0 = ((kt & 1) << 3) + 2 * laneHi;
    const v16h bx = pack16(rp[i0], rp[i0 + 1], rp[i0 + 4], rp[i0 + 5]);
#pragma unroll
    for (int ft = 0; ft < 4; ++ft)
      acc[ft] = wmma16x16x32(load_fragW(sW1p, kt * 4 + ft, lane), bx, acc[ft]);
  }
  // + d2 * W1[128,:] + b1, silu (all per-lane registers)
#pragma unroll
  for (int ft = 0; ft < 4; ++ft) {
    const float4* b4 = (const float4*)&sb1[ft * 16 + 8 * laneHi];
    const float4* w4 = (const float4*)&sW1d2[ft * 16 + 8 * laneHi];
    const float4 b0 = b4[0], b1v = b4[1], w0 = w4[0], w1 = w4[1];
    const float bv[8] = {b0.x, b0.y, b0.z, b0.w, b1v.x, b1v.y, b1v.z, b1v.w};
    const float wv[8] = {w0.x, w0.y, w0.z, w0.w, w1.x, w1.y, w1.z, w1.w};
#pragma unroll
    for (int r = 0; r < 8; ++r)
      acc[ft][r] = silu_f(acc[ft][r] + d2v * wv[r] + bv[r]);
  }

  // GEMM2: D2 = W2^T @ x1^T ; B fragments built in-register from D1
  v8f acc2[4] = {z, z, z, z};
#pragma unroll
  for (int kt = 0; kt < 2; ++kt) {
    const v16h bx = pack16_acc(acc[2 * kt], acc[2 * kt + 1]);
#pragma unroll
    for (int ft = 0; ft < 4; ++ft)
      acc2[ft] =
          wmma16x16x32(load_fragW(sW2p, kt * 4 + ft, lane), bx, acc2[ft]);
  }
#pragma unroll
  for (int ft = 0; ft < 4; ++ft) {
    const float4* b4 = (const float4*)&sb2[ft * 16 + 8 * laneHi];
    const float4 b0 = b4[0], b1v = b4[1];
    const float bv[8] = {b0.x, b0.y, b0.z, b0.w, b1v.x, b1v.y, b1v.z, b1v.w};
#pragma unroll
    for (int r = 0; r < 8; ++r)
      acc2[ft][r] = silu_f(acc2[ft][r] + bv[r]);  // = m^T
  }

  // GEMM3: D3 = Wp1^T @ m^T
  v8f acc3[4] = {z, z, z, z};
#pragma unroll
  for (int kt = 0; kt < 2; ++kt) {
    const v16h bx = pack16_acc(acc2[2 * kt], acc2[2 * kt + 1]);
#pragma unroll
    for (int ft = 0; ft < 4; ++ft)
      acc3[ft] =
          wmma16x16x32(load_fragW(sWp1p, kt * 4 + ft, lane), bx, acc3[ft]);
  }
  // w = silu(D3 + bp1) . Wp2 + bp2 : register dot + one cross-half shuffle
  float partial = 0.0f;
#pragma unroll
  for (int ft = 0; ft < 4; ++ft) {
    const float4* b4 = (const float4*)&sbp1[ft * 16 + 8 * laneHi];
    const float4* w4 = (const float4*)&sWp2[ft * 16 + 8 * laneHi];
    const float4 b0 = b4[0], b1v = b4[1], w0 = w4[0], w1 = w4[1];
    const float bv[8] = {b0.x, b0.y, b0.z, b0.w, b1v.x, b1v.y, b1v.z, b1v.w};
    const float wv[8] = {w0.x, w0.y, w0.z, w0.w, w1.x, w1.y, w1.z, w1.w};
#pragma unroll
    for (int r = 0; r < 8; ++r)
      partial += silu_f(acc3[ft][r] + bv[r]) * wv[r];
  }
  const float w_edge = partial + __shfl_xor(partial, 16, 32) + sbp2;

  // scatter (L2 float atomics): pos_acc/deg once per edge, agg per feature
  if (laneHi == 0) {
    atomicAdd(&pos_acc[d * 2 + 0], r0 * w_edge);
    atomicAdd(&pos_acc[d * 2 + 1], r1 * w_edge);
    atomicAdd(&deg[d], 1.0f);
  }
#pragma unroll
  for (int ft = 0; ft < 4; ++ft)
#pragma unroll
    for (int r = 0; r < 8; ++r)
      atomicAdd(&agg[(size_t)d * DD + ft * 16 + 8 * laneHi + r], acc2[ft][r]);
}

// ---------------------------------------------------------------------------
// node kernel (same transposed chain): phi_h([h, agg]) + residuals + pos
// ---------------------------------------------------------------------------
__global__ __launch_bounds__(256) void egnn_node_kernel(
    const float* __restrict__ h_eff, const float* __restrict__ pos,
    const float* __restrict__ agg, const float* __restrict__ pos_acc,
    const float* __restrict__ deg,
    const float* __restrict__ W_h1, const float* __restrict__ b_h1,
    const float* __restrict__ W_h2, const float* __restrict__ b_h2,
    float* __restrict__ out_h, float* __restrict__ out_pos) {
  __shared__ __align__(32) _Float16 sWh1p[16 * 32 * 16];
  __shared__ __align__(32) _Float16 sWh2p[8 * 32 * 16];
  __shared__ __align__(16) float sbh1[DD], sbh2[DD];

  const int tid = threadIdx.x;
  pack_weight(W_h1, sWh1p, 4, tid, 256);
  pack_weight(W_h2, sWh2p, 2, tid, 256);
  if (tid < DD) { sbh1[tid] = b_h1[tid]; sbh2[tid] = b_h2[tid]; }
  __syncthreads();

  const int wave = tid >> 5, lane = tid & 31;
  const int laneHi = lane >> 4, lr = lane & 15;
  const int node = blockIdx.x * 128 + wave * 16 + lr;
  const bool ok = node < NN;
  const int nsafe = ok ? node : 0;
  const float* __restrict__ hrow = &h_eff[(size_t)nsafe * DD];
  const float* __restrict__ arow = &agg[(size_t)nsafe * DD];

  const v8f z = {};
  float4 hold[8];  // this lane's 32 h_eff values (reused for residual)

  // GEMM1: D1 = Wh1^T @ [h|agg]^T
  v8f acc[4] = {z, z, z, z};
#pragma unroll
  for (int kt = 0; kt < 4; ++kt) {
    const float4* rp = (const float4*)((kt < 2) ? hrow : arow);
    const int i0 = ((kt & 1) << 3) + 2 * laneHi;
    const float4 a = rp[i0], b = rp[i0 + 1], c = rp[i0 + 4], e = rp[i0 + 5];
    if (kt < 2) {
      hold[kt * 4 + 0] = a; hold[kt * 4 + 1] = b;
      hold[kt * 4 + 2] = c; hold[kt * 4 + 3] = e;
    }
    const v16h bx = pack16(a, b, c, e);
#pragma unroll
    for (int ft = 0; ft < 4; ++ft)
      acc[ft] = wmma16x16x32(load_fragW(sWh1p, kt * 4 + ft, lane), bx, acc[ft]);
  }
#pragma unroll
  for (int ft = 0; ft < 4; ++ft) {
    const float4* b4 = (const float4*)&sbh1[ft * 16 + 8 * laneHi];
    const float4 b0 = b4[0], b1v = b4[1];
    const float bv[8] = {b0.x, b0.y, b0.z, b0.w, b1v.x, b1v.y, b1v.z, b1v.w};
#pragma unroll
    for (int r = 0; r < 8; ++r)
      acc[ft][r] = silu_f(acc[ft][r] + bv[r]);
  }

  // GEMM2: D2 = Wh2^T @ x^T
  v8f acc2[4] = {z, z, z, z};
#pragma unroll
  for (int kt = 0; kt < 2; ++kt) {
    const v16h bx = pack16_acc(acc[2 * kt], acc[2 * kt + 1]);
#pragma unroll
    for (int ft = 0; ft < 4; ++ft)
      acc2[ft] =
          wmma16x16x32(load_fragW(sWh2p, kt * 4 + ft, lane), bx, acc2[ft]);
  }

  if (ok) {
    // out_h = h_eff + (D2 + b_h2), vectorized float4 stores
#pragma unroll
    for (int ft = 0; ft < 4; ++ft) {
      const float4* b4 = (const float4*)&sbh2[ft * 16 + 8 * laneHi];
      const float4 b0 = b4[0], b1v = b4[1];
      const float4 h0 = hold[2 * ft], h1 = hold[2 * ft + 1];
      float4 o0, o1;
      o0.x = h0.x + acc2[ft][0] + b0.x;
      o0.y = h0.y + acc2[ft][1] + b0.y;
      o0.z = h0.z + acc2[ft][2] + b0.z;
      o0.w = h0.w + acc2[ft][3] + b0.w;
      o1.x = h1.x + acc2[ft][4] + b1v.x;
      o1.y = h1.y + acc2[ft][5] + b1v.y;
      o1.z = h1.z + acc2[ft][6] + b1v.z;
      o1.w = h1.w + acc2[ft][7] + b1v.w;
      float* op = &out_h[(size_t)node * DD + ft * 16 + 8 * laneHi];
      *(float4*)&op[0] = o0;
      *(float4*)&op[4] = o1;
    }
    if (laneHi == 0) {
      const float dg = fmaxf(deg[node], 1.0f);
      out_pos[node * 2 + 0] = pos[node * 2 + 0] + pos_acc[node * 2 + 0] / dg;
      out_pos[node * 2 + 1] = pos[node * 2 + 1] + pos_acc[node * 2 + 1] / dg;
    }
  }
}

// ---------------------------------------------------------------------------
extern "C" void kernel_launch(void* const* d_in, const int* in_sizes, int n_in,
                              void* d_out, int out_size, void* d_ws,
                              size_t ws_size, hipStream_t stream) {
  const float* h      = (const float*)d_in[0];
  const float* pos    = (const float*)d_in[1];
  const float* h_init = (const float*)d_in[2];
  const float* W_msg1 = (const float*)d_in[3];
  const float* b_msg1 = (const float*)d_in[4];
  const float* W_msg2 = (const float*)d_in[5];
  const float* b_msg2 = (const float*)d_in[6];
  const float* W_pos1 = (const float*)d_in[7];
  const float* b_pos1 = (const float*)d_in[8];
  const float* W_pos2 = (const float*)d_in[9];
  const float* b_pos2 = (const float*)d_in[10];
  const float* W_h1   = (const float*)d_in[11];
  const float* b_h1   = (const float*)d_in[12];
  const float* W_h2   = (const float*)d_in[13];
  const float* b_h2   = (const float*)d_in[14];
  const int*   ei     = (const int*)d_in[15];

  // workspace layout (f32): h_eff[N*64] | agg[N*64] | pos_acc[N*2] | deg[N]
  float* h_eff   = (float*)d_ws;
  float* agg     = h_eff + (size_t)NN * DD;
  float* pos_acc = agg + (size_t)NN * DD;
  float* deg     = pos_acc + (size_t)NN * 2;

  float* out_h   = (float*)d_out;             // [N,64]
  float* out_pos = out_h + (size_t)NN * DD;   // [N,2]

  egnn_prep_kernel<<<(NN * DD + 255) / 256, 256, 0, stream>>>(
      h, h_init, h_eff, agg, pos_acc, deg);

  egnn_edge_kernel<<<EE / 128, 256, 0, stream>>>(
      h_eff, pos, ei, W_msg1, b_msg1, W_msg2, b_msg2, W_pos1, b_pos1, W_pos2,
      b_pos2, agg, pos_acc, deg);

  egnn_node_kernel<<<(NN + 127) / 128, 256, 0, stream>>>(
      h_eff, pos, agg, pos_acc, deg, W_h1, b_h1, W_h2, b_h2, out_h, out_pos);
}
